// SNNQNet_7782480740671
// MI455X (gfx1250) — matrix operand order repaired and checked by reading, not compiled, not run on hardware
//
#include <hip/hip_runtime.h>

// ---------------------------------------------------------------------------
// SNN Q-network for MI455X (gfx1250, wave32, WMMA).
// B=16384, S=256, H=128, A=18, T=8.
// All matmul A-operands are binary spikes -> exact in f16.
// Weights split into f16 hi + f16 lo (two WMMAs into one f32 accumulator)
// for ~f32 accuracy on the fast 16x16x32 f16 WMMA pipe.
// Weight panels live in LDS (loaded once per workgroup); a __syncthreads()
// at the top of each timestep pins the ds_load fragment reads inside the
// t-loop so the compiler cannot hoist+spill them (round-1 failure mode).
// Per wave: 16 batch rows x (64|128) cols x 8 timesteps with LIF membrane
// state in registers; matmul + bias + LN + residual + LIF + spike store
// fused per layer. Spike sequences ping-pong in workspace (f16, 33.5 MB
// each -> L2-resident at 192 MB L2).
// ---------------------------------------------------------------------------

typedef __attribute__((ext_vector_type(16))) _Float16 v16h;
typedef __attribute__((ext_vector_type(8)))  float    v8f;

#define TT   8
#define BB   16384
#define SS   256
#define HH   128
#define AO   18
#define APAD 32

union AFrag {
  v16h     v;
  _Float16 h[16];
  uint4    q[2];
};

__device__ __forceinline__ v8f wmma_f16(v16h a, v16h b, v8f c) {
  // D = A(16x32 f16) * B(32x16 f16) + C(16x16 f32)
  return __builtin_amdgcn_wmma_f32_16x16x32_f16(
      /*neg_a=*/false, a, /*neg_b=*/false, b,
      /*c_mod=*/(short)0, c, /*reuse_a=*/false, /*reuse_b=*/false);
}

__device__ __forceinline__ float hred16(float p) {
  // reduce across the 16-lane half-group (xor 1,2,4,8 stays within halves)
  p += __shfl_xor(p, 1, 32);
  p += __shfl_xor(p, 2, 32);
  p += __shfl_xor(p, 4, 32);
  p += __shfl_xor(p, 8, 32);
  return p;
}

// ---------------------------------------------------------------------------
// Weight prepack: W[K][Nreal] f32 -> B-fragment panels, hi/lo f16 split.
// Fragment (kt,ct): 512 halves; lane L = 16*((k/8)%2)+n%16, half h ->
// k = kt*32 + 8*(L/16) + (h%8) + 16*(h/8). Lane's 16 halves contiguous (32B).
// ---------------------------------------------------------------------------
__global__ void pack_panel(const float* __restrict__ W, int K, int Nreal, int Npad,
                           _Float16* __restrict__ hi, _Float16* __restrict__ lo) {
  int idx = blockIdx.x * blockDim.x + threadIdx.x;
  if (idx >= K * Npad) return;
  int k = idx / Npad, n = idx % Npad;
  float w = (n < Nreal) ? W[k * Nreal + n] : 0.0f;
  _Float16 whi = (_Float16)w;
  _Float16 wlo = (_Float16)(w - (float)whi);
  int kt = k >> 5, kr = k & 31, g = (kr >> 3) & 1, h = (kr & 7) | ((kr >> 4) << 3);
  int ct = n >> 4, nl = n & 15;
  int CT = Npad >> 4;
  int off = (((kt * CT + ct) << 9) | ((g * 16 + nl) << 4)) + h;
  hi[off] = whi;
  lo[off] = wlo;
}

// ---------------------------------------------------------------------------
// Phase 1: poisson-encode -> [B,S]x[S,H] matmul (hi/lo WMMA) -> LIF -> spikes
// N split across gridDim.y (no LN here -> columns independent).
// Wave: 16 rows x 64 cols x 8 t.  LDS: 256x64 hi+lo panel = 64 KB.
// ---------------------------------------------------------------------------
__global__ void __launch_bounds__(256, 1)
snn_phase1(const float* __restrict__ x, const float* __restrict__ enc,
           const _Float16* __restrict__ whi, const _Float16* __restrict__ wlo,
           const float* __restrict__ bias, _Float16* __restrict__ sp_out) {
  __shared__ __align__(32) _Float16 lds_w[2][32 * 512];  // 32 frags hi + lo

  const int lane    = threadIdx.x & 31;
  const int wave    = threadIdx.x >> 5;
  const int rowbase = (blockIdx.x * 8 + wave) * 16;
  const int cg      = blockIdx.y;      // column group (0/1): cols cg*64..+63
  const int nl      = lane & 15;       // C-layout column within tile
  const int mg      = lane >> 4;       // C-layout row group
  const int rowA    = rowbase + nl;    // A-layout row = lane%16
  const int ka      = (lane >> 4) * 8; // A-layout k sub-offset per half-group

  // Cooperative panel load: local frag f = kt*4+c  <->  global frag kt*8+cg*4+c
  {
    const uint4* gh = (const uint4*)whi;
    const uint4* gl = (const uint4*)wlo;
    uint4* lh = (uint4*)&lds_w[0][0];
    uint4* ll = (uint4*)&lds_w[1][0];
    for (int i = threadIdx.x; i < 2048; i += 256) {   // 64 uint4 per frag
      int f = i >> 6, w = i & 63;
      int gf = ((f >> 2) << 3) + (cg << 2) + (f & 3);
      lh[i] = gh[(gf << 6) + w];
      ll[i] = gl[(gf << 6) + w];
    }
  }

  float bs[4];
#pragma unroll
  for (int c = 0; c < 4; c++) bs[c] = bias[cg * 64 + c * 16 + nl];

  float v[4][8];
#pragma unroll
  for (int c = 0; c < 4; c++)
#pragma unroll
    for (int r = 0; r < 8; r++) v[c][r] = 0.0f;

  const float* xp = x + (long)rowA * SS;

  for (int t = 0; t < TT; t++) {
    __syncthreads();   // fence: keeps ds_load fragments inside the loop

    v8f acc[4];
#pragma unroll
    for (int c = 0; c < 4; c++)
#pragma unroll
      for (int r = 0; r < 8; r++) acc[c][r] = 0.0f;

    const float* ep = enc + ((long)t * BB + rowA) * SS;

    for (int kt = 0; kt < 8; kt++) {   // K = 256 -> 8 k-tiles of 32
      const int kb = kt * 32 + ka;
      AFrag a;
#pragma unroll
      for (int i = 0; i < 8; i++)
        a.h[i] = (ep[kb + i] <= xp[kb + i]) ? (_Float16)1.0f : (_Float16)0.0f;
#pragma unroll
      for (int i = 0; i < 8; i++)
        a.h[8 + i] = (ep[kb + 16 + i] <= xp[kb + 16 + i]) ? (_Float16)1.0f
                                                          : (_Float16)0.0f;
#pragma unroll
      for (int c = 0; c < 4; c++) {
        const int fo = ((kt * 4 + c) << 9) + (lane << 4);
        v16h bh = *(const v16h*)&lds_w[0][fo];
        v16h bl = *(const v16h*)&lds_w[1][fo];
        acc[c] = wmma_f16(a.v, bh, acc[c]);
        acc[c] = wmma_f16(a.v, bl, acc[c]);
      }
    }

    // LIF: v += (z - v)/2; s = H(v-1); hard reset
#pragma unroll
    for (int c = 0; c < 4; c++) {
#pragma unroll
      for (int r = 0; r < 8; r++) {
        float z  = acc[c][r] + bs[c];
        float vv = v[c][r] + 0.5f * (z - v[c][r]);
        float s  = (vv >= 1.0f) ? 1.0f : 0.0f;
        v[c][r]  = vv * (1.0f - s);
        sp_out[((long)t * BB + rowbase + r + 8 * mg) * HH + cg * 64 +
               c * 16 + nl] = (_Float16)s;
      }
    }
  }
}

// ---------------------------------------------------------------------------
// Residual block: spikes -> [.,H]x[H,H] matmul -> +bias -> LN -> +h -> LIF
// Wave: 16 rows x 128 cols x 8 t.  LDS: 128x128 hi+lo panel = 64 KB.
// ---------------------------------------------------------------------------
__global__ void __launch_bounds__(256, 1)
snn_block(const _Float16* __restrict__ sp_in,
          const _Float16* __restrict__ whi, const _Float16* __restrict__ wlo,
          const float* __restrict__ bias, const float* __restrict__ gamma,
          const float* __restrict__ beta, _Float16* __restrict__ sp_out) {
  __shared__ __align__(32) _Float16 lds_w[2][32 * 512];  // 32 frags hi + lo

  const int lane    = threadIdx.x & 31;
  const int wave    = threadIdx.x >> 5;
  const int rowbase = (blockIdx.x * 8 + wave) * 16;
  const int nl      = lane & 15;
  const int mg      = lane >> 4;
  const int rowA    = rowbase + nl;
  const int ka      = (lane >> 4) * 8;

  {
    const uint4* gh = (const uint4*)whi;
    const uint4* gl = (const uint4*)wlo;
    uint4* lh = (uint4*)&lds_w[0][0];
    uint4* ll = (uint4*)&lds_w[1][0];
    for (int i = threadIdx.x; i < 2048; i += 256) {
      lh[i] = gh[i];
      ll[i] = gl[i];
    }
  }

  float bs[8], gm[8], bt[8];
#pragma unroll
  for (int c = 0; c < 8; c++) {
    int n = c * 16 + nl;
    bs[c] = bias[n];
    gm[c] = gamma[n];
    bt[c] = beta[n];
  }

  float v[8][8];
#pragma unroll
  for (int c = 0; c < 8; c++)
#pragma unroll
    for (int r = 0; r < 8; r++) v[c][r] = 0.0f;

  for (int t = 0; t < TT; t++) {
    __syncthreads();   // fence: keeps ds_load fragments inside the loop

    v8f acc[8];
#pragma unroll
    for (int c = 0; c < 8; c++)
#pragma unroll
      for (int r = 0; r < 8; r++) acc[c][r] = 0.0f;

    const _Float16* arow = sp_in + ((long)t * BB + rowA) * HH;

    for (int kt = 0; kt < 4; kt++) {   // K = 128 -> 4 k-tiles of 32
      const int kb = kt * 32 + ka;
      AFrag a;
      const uint4* sp = (const uint4*)(arow + kb);
      a.q[0] = sp[0];   // halves kb .. kb+7
      a.q[1] = sp[2];   // halves kb+16 .. kb+23
#pragma unroll
      for (int c = 0; c < 8; c++) {
        const int fo = ((kt * 8 + c) << 9) + (lane << 4);
        v16h bh = *(const v16h*)&lds_w[0][fo];
        v16h bl = *(const v16h*)&lds_w[1][fo];
        acc[c] = wmma_f16(a.v, bh, acc[c]);
        acc[c] = wmma_f16(a.v, bl, acc[c]);
      }
    }

    // LayerNorm stats per row (over 128 features)
    float mu[8], rs[8];
#pragma unroll
    for (int r = 0; r < 8; r++) {
      float p = 0.0f, q = 0.0f;
#pragma unroll
      for (int c = 0; c < 8; c++) {
        float val = acc[c][r] + bs[c];
        p += val;
        q += val * val;
      }
      p = hred16(p);
      q = hred16(q);
      float m  = p * (1.0f / 128.0f);
      float va = q * (1.0f / 128.0f) - m * m;
      mu[r] = m;
      rs[r] = rsqrtf(va + 1e-5f);
    }

    // normalize + residual + LIF
#pragma unroll
    for (int c = 0; c < 8; c++) {
#pragma unroll
      for (int r = 0; r < 8; r++) {
        long  idx = ((long)t * BB + rowbase + r + 8 * mg) * HH + c * 16 + nl;
        float val = acc[c][r] + bs[c];
        float ln  = (val - mu[r]) * rs[r] * gm[c] + bt[c];
        float zz  = ln + (float)sp_in[idx];
        float vv  = v[c][r] + 0.5f * (zz - v[c][r]);
        float s   = (vv >= 1.0f) ? 1.0f : 0.0f;
        v[c][r]   = vv * (1.0f - s);
        sp_out[idx] = (_Float16)s;
      }
    }
  }
}

// ---------------------------------------------------------------------------
// Output: spikes -> [.,H]x[H,18(pad32)] -> non-spiking LIF -> mean over T
// Wave: 16 rows x 32 cols x 8 t.  LDS: 128x32 hi+lo panel = 16 KB.
// ---------------------------------------------------------------------------
__global__ void __launch_bounds__(256, 1)
snn_out(const _Float16* __restrict__ sp_in,
        const _Float16* __restrict__ whi, const _Float16* __restrict__ wlo,
        const float* __restrict__ bout, float* __restrict__ out) {
  __shared__ __align__(32) _Float16 lds_w[2][8 * 512];   // 8 frags hi + lo

  const int lane    = threadIdx.x & 31;
  const int wave    = threadIdx.x >> 5;
  const int rowbase = (blockIdx.x * 8 + wave) * 16;
  const int nl      = lane & 15;
  const int mg      = lane >> 4;
  const int rowA    = rowbase + nl;
  const int ka      = (lane >> 4) * 8;

  {
    const uint4* gh = (const uint4*)whi;
    const uint4* gl = (const uint4*)wlo;
    uint4* lh = (uint4*)&lds_w[0][0];
    uint4* ll = (uint4*)&lds_w[1][0];
    for (int i = threadIdx.x; i < 512; i += 256) {
      lh[i] = gh[i];
      ll[i] = gl[i];
    }
  }

  float bs[2];
#pragma unroll
  for (int c = 0; c < 2; c++) {
    int n = c * 16 + nl;
    bs[c] = (n < AO) ? bout[n] : 0.0f;
  }

  float v[2][8], sm[2][8];
#pragma unroll
  for (int c = 0; c < 2; c++)
#pragma unroll
    for (int r = 0; r < 8; r++) { v[c][r] = 0.0f; sm[c][r] = 0.0f; }

  for (int t = 0; t < TT; t++) {
    __syncthreads();   // fence: keeps ds_load fragments inside the loop

    v8f acc[2];
#pragma unroll
    for (int c = 0; c < 2; c++)
#pragma unroll
      for (int r = 0; r < 8; r++) acc[c][r] = 0.0f;

    const _Float16* arow = sp_in + ((long)t * BB + rowA) * HH;

    for (int kt = 0; kt < 4; kt++) {
      const int kb = kt * 32 + ka;
      AFrag a;
      const uint4* sp = (const uint4*)(arow + kb);
      a.q[0] = sp[0];
      a.q[1] = sp[2];
#pragma unroll
      for (int c = 0; c < 2; c++) {
        const int fo = ((kt * 2 + c) << 9) + (lane << 4);
        v16h bh = *(const v16h*)&lds_w[0][fo];
        v16h bl = *(const v16h*)&lds_w[1][fo];
        acc[c] = wmma_f16(a.v, bh, acc[c]);
        acc[c] = wmma_f16(a.v, bl, acc[c]);
      }
    }

#pragma unroll
    for (int c = 0; c < 2; c++) {
#pragma unroll
      for (int r = 0; r < 8; r++) {
        float z  = acc[c][r] + bs[c];
        float vv = v[c][r] + 0.5f * (z - v[c][r]);
        v[c][r]  = vv;
        sm[c][r] += vv;
      }
    }
  }

#pragma unroll
  for (int c = 0; c < 2; c++) {
#pragma unroll
    for (int r = 0; r < 8; r++) {
      int n = c * 16 + nl;
      if (n < AO)
        out[(long)(rowbase + r + 8 * mg) * AO + n] = sm[c][r] * 0.125f;
    }
  }
}

// ---------------------------------------------------------------------------
extern "C" void kernel_launch(void* const* d_in, const int* in_sizes, int n_in,
                              void* d_out, int out_size, void* d_ws, size_t ws_size,
                              hipStream_t stream) {
  const float* x     = (const float*)d_in[0];
  const float* enc   = (const float*)d_in[1];
  const float* W_in  = (const float*)d_in[2];
  const float* b_in  = (const float*)d_in[3];
  const float* W_r1  = (const float*)d_in[4];
  const float* b_r1  = (const float*)d_in[5];
  const float* g_r1  = (const float*)d_in[6];
  const float* be_r1 = (const float*)d_in[7];
  const float* W_r2  = (const float*)d_in[8];
  const float* b_r2  = (const float*)d_in[9];
  const float* g_r2  = (const float*)d_in[10];
  const float* be_r2 = (const float*)d_in[11];
  const float* W_out = (const float*)d_in[12];
  const float* b_out = (const float*)d_in[13];
  float* out = (float*)d_out;

  char* ws = (char*)d_ws;
  const size_t SPIKE_BYTES = (size_t)TT * BB * HH * sizeof(_Float16); // 33.5 MB
  _Float16* spike0 = (_Float16*)ws;
  _Float16* spike1 = (_Float16*)(ws + SPIKE_BYTES);
  _Float16* pan    = (_Float16*)(ws + 2 * SPIKE_BYTES);
  _Float16* win_hi  = pan;               // 256*128
  _Float16* win_lo  = win_hi + 32768;
  _Float16* wr1_hi  = win_lo + 32768;    // 128*128
  _Float16* wr1_lo  = wr1_hi + 16384;
  _Float16* wr2_hi  = wr1_lo + 16384;
  _Float16* wr2_lo  = wr2_hi + 16384;
  _Float16* wout_hi = wr2_lo + 16384;    // 128*32 (padded)
  _Float16* wout_lo = wout_hi + 4096;

  // Prepack weights into WMMA B-fragment panels (hi/lo f16 split)
  pack_panel<<<(SS * HH + 255) / 256, 256, 0, stream>>>(W_in, SS, HH, HH, win_hi, win_lo);
  pack_panel<<<(HH * HH + 255) / 256, 256, 0, stream>>>(W_r1, HH, HH, HH, wr1_hi, wr1_lo);
  pack_panel<<<(HH * HH + 255) / 256, 256, 0, stream>>>(W_r2, HH, HH, HH, wr2_hi, wr2_lo);
  pack_panel<<<(HH * APAD + 255) / 256, 256, 0, stream>>>(W_out, HH, AO, APAD, wout_hi, wout_lo);

  snn_phase1<<<dim3(BB / 128, 2), 256, 0, stream>>>(x, enc, win_hi, win_lo, b_in, spike0);
  snn_block <<<BB / 128, 256, 0, stream>>>(spike0, wr1_hi, wr1_lo, b_r1, g_r1, be_r1, spike1);
  snn_block <<<BB / 128, 256, 0, stream>>>(spike1, wr2_hi, wr2_lo, b_r2, g_r2, be_r2, spike0);
  snn_out   <<<BB / 128, 256, 0, stream>>>(spike0, wout_hi, wout_lo, b_out, out);
}
